// TripletMPNN_63943473103173
// MI455X (gfx1250) — compile-verified, matrix operand order checked by reading, not run on hardware
//
#include <hip/hip_runtime.h>

#define DEVINL static __device__ __forceinline__

typedef _Float16 h16 __attribute__((ext_vector_type(16)));
typedef float f32x8 __attribute__((ext_vector_type(8)));
typedef unsigned int u32x4 __attribute__((ext_vector_type(4)));
typedef int i32x4 __attribute__((ext_vector_type(4)));
typedef int i32x8 __attribute__((ext_vector_type(8)));

constexpr int Bb = 4, Nn = 128, Cc = 256, Ee = 128, Oo = 128;
constexpr float NEGV = -1.0e9f;

#if defined(__gfx1250__) && __has_builtin(__builtin_amdgcn_tensor_load_to_lds) && \
    __has_builtin(__builtin_amdgcn_s_wait_tensorcnt)
#define USE_TDM 1
#endif

// ---------------- WMMA helpers (gfx1250, wave32) ----------------
// A fragment (16x32 f16): lane holds row M=lane&15; K-halves selected by lane>>4.
// B fragment: weights pre-transposed to [col][K] so B loads mirror A loads.
// C/D (16x16 f32): lane holds col N=lane&15; VGPR v holds row M=v+(lane>>4)*8.

DEVINL f32x8 wmma16(h16 a, h16 b, f32x8 c) {
  return __builtin_amdgcn_wmma_f32_16x16x32_f16(false, a, false, b, (short)0, c,
                                                false, false);
}

DEVINL h16 loadA_f32(const float* p) {  // p = row base + kstep*32 + (lane>>4)*8
  h16 r;
#pragma unroll
  for (int v = 0; v < 8; ++v) {
    r[v] = (_Float16)p[v];
    r[8 + v] = (_Float16)p[16 + v];
  }
  return r;
}

DEVINL h16 loadB_h(const _Float16* p) {  // p = col base + kstep*32 + (lane>>4)*8
  h16 r;
#pragma unroll
  for (int v = 0; v < 8; ++v) {
    r[v] = p[v];
    r[8 + v] = p[16 + v];
  }
  return r;
}

DEVINL void atomicMaxF(float* addr, float val) {
#if defined(__gfx1250__)
  // No-return f32 max atomic at device scope (tracked on STOREcnt; S_ENDPGM's
  // implicit wait-idle guarantees completion before the next kernel).
  asm volatile("global_atomic_max_num_f32 %0, %1, off scope:SCOPE_DEV"
               :
               : "v"((unsigned long long)(uintptr_t)addr), "v"(val)
               : "memory");
#else
  unsigned int* ua = (unsigned int*)addr;
  unsigned int old = *ua;
  while (__uint_as_float(old) < val) {
    unsigned int assumed = old;
    old = atomicCAS(ua, assumed, __float_as_uint(val));
    if (old == assumed) break;
  }
#endif
}

// ---------------- K0: weight convert/transpose + msgs init ----------------
__global__ __launch_bounds__(256) void k_prep(
    const float* Wme, const float* Wmlp1, const float* Wmlp2, const float* WU2,
    const float* Wm1, const float* Wm2, const float* WU1,
    const float* Wte1, const float* Wte2, const float* Wte3,
    _Float16* WmeT, _Float16* Wmlp1T, _Float16* Wmlp2T, _Float16* WU2T,
    _Float16* Wm1T, _Float16* Wm2T, _Float16* WU1T, _Float16* WteCT,
    float* msgs_ws) {
  int idx = blockIdx.x * blockDim.x + threadIdx.x;
  if (idx < 4 * 16384) {  // 128x128 -> T
    int w = idx >> 14, r = idx & 16383, o = r >> 7, k = r & 127;
    const float* src = (w == 0) ? Wme : (w == 1) ? Wmlp1 : (w == 2) ? Wmlp2 : WU2;
    _Float16* dst = (w == 0) ? WmeT : (w == 1) ? Wmlp1T : (w == 2) ? Wmlp2T : WU2T;
    dst[o * 128 + k] = (_Float16)src[k * 128 + o];
    return;
  }
  idx -= 4 * 16384;
  if (idx < 3 * 32768) {  // 256x128 -> T (128x256)
    int w = idx >> 15, r = idx & 32767, o = r >> 8, k = r & 255;
    const float* src = (w == 0) ? Wm1 : (w == 1) ? Wm2 : WU1;
    _Float16* dst = (w == 0) ? Wm1T : (w == 1) ? Wm2T : WU1T;
    dst[o * 256 + k] = (_Float16)src[k * 128 + o];
    return;
  }
  idx -= 3 * 32768;
  if (idx < 32 * 128) {  // combined te weights, cols 0..23 valid, 24..31 zero
    int n = idx >> 7, k = idx & 127;
    float v = 0.f;
    if (n < 8) v = Wte1[k * 8 + n];
    else if (n < 16) v = Wte2[k * 8 + (n - 8)];
    else if (n < 24) v = Wte3[k * 8 + (n - 16)];
    WteCT[n * 128 + k] = (_Float16)v;
    return;
  }
  idx -= 32 * 128;
  if (idx < Bb * Nn * Oo) msgs_ws[idx] = NEGV;  // identity for max-reduce
}

// ---------------- K1: small scalar dots (t1..t3, tg, mg) ----------------
__global__ __launch_bounds__(256) void k_small(
    const float* z, const float* g, const unsigned char* mask,
    const float* Wt1, const float* Wt2, const float* Wt3,
    const float* Wtg, const float* Wmg,
    float* t1, float* t2, float* t3, float* tg, float* mg) {
  int idx = blockIdx.x * blockDim.x + threadIdx.x;
  const int NT = Bb * Nn * 8;  // 4096
  if (idx < 3 * NT) {
    int w = idx / NT, r = idx % NT, bn = r >> 3, h = r & 7;
    const float* W = (w == 0) ? Wt1 : (w == 1) ? Wt2 : Wt3;
    const float* zr = z + (size_t)bn * Cc;
    float s = 0.f;
#pragma unroll 4
    for (int k = 0; k < Cc; ++k) s += zr[k] * W[k * 8 + h];
    float val = mask[bn] ? s : 0.f;
    ((w == 0) ? t1 : (w == 1) ? t2 : t3)[bn * 8 + h] = val;
    return;
  }
  idx -= 3 * NT;
  if (idx < Bb * 8) {
    int b = idx >> 3, h = idx & 7;
    float s = 0.f;
    for (int k = 0; k < Cc / 2; ++k) s += g[b * 128 + k] * Wtg[k * 8 + h];
    tg[b * 8 + h] = s;
    return;
  }
  idx -= Bb * 8;
  if (idx < Bb * Oo) {
    int b = idx >> 7, o = idx & 127;
    float s = 0.f;
    for (int k = 0; k < Cc / 2; ++k) s += g[b * 128 + k] * Wmg[k * 128 + o];
    mg[b * 128 + o] = s;
  }
}

// ---------------- K2: m1 = mask*(z@Wm1), m2 = mask*(z@Wm2) via WMMA ----------
__global__ __launch_bounds__(32) void k_m12(
    const float* z, const unsigned char* mask,
    const _Float16* Wm1T, const _Float16* Wm2T, float* m1, float* m2) {
  int which = blockIdx.x >> 5;  // 0 -> m1, 1 -> m2
  int tile = blockIdx.x & 31;   // 32 tiles of 16 rows over B*N=512
  const _Float16* WT = which ? Wm2T : Wm1T;
  float* outp = which ? m2 : m1;
  int lane = threadIdx.x, n = lane & 15, hs = lane >> 4, koff = hs * 8;
  const float* zrow = z + ((size_t)(tile * 16 + n)) * Cc;
  f32x8 acc[8];
#pragma unroll
  for (int nt = 0; nt < 8; ++nt) acc[nt] = (f32x8)0.0f;
#pragma unroll
  for (int ks = 0; ks < 8; ++ks) {
    h16 a = loadA_f32(zrow + ks * 32 + koff);
#pragma unroll
    for (int nt = 0; nt < 8; ++nt)
      acc[nt] = wmma16(a, loadB_h(WT + (nt * 16 + n) * Cc + ks * 32 + koff), acc[nt]);
  }
#pragma unroll
  for (int nt = 0; nt < 8; ++nt)
#pragma unroll
    for (int v = 0; v < 8; ++v) {
      int r = tile * 16 + v + hs * 8, o = nt * 16 + n;
      outp[(size_t)r * Oo + o] = mask[r] ? acc[nt][v] : 0.f;
    }
}

// ---------------- K3: fused te-GEMM + me-GEMM + MLP + max-reduce -------------
__global__ __launch_bounds__(32) void k_pair(
    const float* e, const unsigned char* mmask,
    const _Float16* WmeT, const _Float16* Wmlp1T, const _Float16* Wmlp2T,
    const _Float16* WteCT, const float* m1, const float* m2, const float* mg,
    float* te1, float* te2, float* te3, float* msgs_ws) {
  __shared__ float elds[16 * 128];  // TDM staging for the 8KB e-slab
  __shared__ float lds[16 * 128];   // activation staging between GEMMs
  int t = blockIdx.x;               // 4096 tiles: (b, i, j0/16)
  int b = t >> 10, rem = t & 1023, i = rem >> 3, j0 = (rem & 7) << 4;
  int lane = threadIdx.x, n = lane & 15, hs = lane >> 4, koff = hs * 8;

#if defined(USE_TDM)
  {
    // D# for a 1-D contiguous tile: rows (b,i,j0..j0+15) of e are adjacent,
    // so the tile is 2048 consecutive f32 (8KB) -> LDS.
    unsigned long long ga =
        (unsigned long long)(uintptr_t)(e + ((size_t)(b * Nn + i) * Nn + j0) * Ee);
    unsigned lof = (unsigned)(uintptr_t)(void*)elds;  // addr[31:0] = LDS offset
    u32x4 g0;
    g0[0] = 1u;                                       // count=1 (valid user D#)
    g0[1] = lof;                                      // lds_addr
    g0[2] = (unsigned)(ga & 0xffffffffu);             // global_addr[31:0]
    g0[3] = (unsigned)((ga >> 32) & 0x1ffffffu) | (2u << 30);  // [56:32] | type=2
    i32x8 g1;
    g1[0] = 0x00020000;          // workgroup_mask=0, data_size=2 (4B)
    g1[1] = (int)(2048u << 16);  // tensor_dim0[15:0]=2048 (bits 63:48)
    g1[2] = (int)(1u << 16);     // tensor_dim0 hi=0, tensor_dim1[15:0]=1
    g1[3] = (int)(2048u << 16);  // tensor_dim1 hi=0, tile_dim0=2048
    g1[4] = 0;                   // tile_dim1=0 (1-D), tile_dim2=0
    g1[5] = 2048;                // tensor_dim0_stride[31:0]
    g1[6] = 0;                   // stride hi / dim1_stride lo
    g1[7] = 0;
#if __clang_major__ >= 23
    __builtin_amdgcn_tensor_load_to_lds(g0, g1, (i32x4)0, (i32x4)0, (i32x8)0, 0);
#else
    __builtin_amdgcn_tensor_load_to_lds(g0, g1, (i32x4)0, (i32x4)0, 0);
#endif
    __builtin_amdgcn_s_wait_tensorcnt(0);
  }
  const float* arow = &elds[(lane & 15) * 128];  // A row for this lane (LDS)
#else
  const float* arow = e + (((size_t)(b * Nn + i) * Nn) + (j0 + n)) * Ee;
  __builtin_prefetch(arow, 0, 3);
  __builtin_prefetch(arow + 64, 0, 3);
#endif

  f32x8 accme[8], accte[2];
#pragma unroll
  for (int nt = 0; nt < 8; ++nt) accme[nt] = (f32x8)0.0f;
  accte[0] = (f32x8)0.0f;
  accte[1] = (f32x8)0.0f;

#pragma unroll
  for (int ks = 0; ks < 4; ++ks) {
    h16 a = loadA_f32(arow + ks * 32 + koff);
#pragma unroll
    for (int nt = 0; nt < 2; ++nt)
      accte[nt] = wmma16(a, loadB_h(WteCT + (nt * 16 + n) * Ee + ks * 32 + koff), accte[nt]);
#pragma unroll
    for (int nt = 0; nt < 8; ++nt)
      accme[nt] = wmma16(a, loadB_h(WmeT + (nt * 16 + n) * Ee + ks * 32 + koff), accme[nt]);
  }

  // scatter te results: ntile0 cols 0-7 -> te1, 8-15 -> te2; ntile1 cols 0-7 -> te3
#pragma unroll
  for (int v = 0; v < 8; ++v) {
    int m = v + hs * 8, j = j0 + m;
    size_t rb = ((size_t)(b * Nn + i) * Nn + j) * 8;
    if (n < 8) {
      te1[rb + n] = accte[0][v];
      te3[rb + n] = accte[1][v];
    } else {
      te2[rb + (n - 8)] = accte[0][v];
    }
  }

  unsigned char mv[8];
#pragma unroll
  for (int v = 0; v < 8; ++v)
    mv[v] = mmask[(b * Nn + i) * Nn + (j0 + v + hs * 8)];

  // bias + ReLU -> LDS staging (C-layout -> row-major tile)
#pragma unroll
  for (int nt = 0; nt < 8; ++nt) {
    int o = nt * 16 + n;
    float b2 = m2[(size_t)(b * Nn + i) * Oo + o] + mg[b * Oo + o];
#pragma unroll
    for (int v = 0; v < 8; ++v) {
      int m = v + hs * 8, j = j0 + m;
      float x = (mv[v] ? accme[nt][v] : 0.f) + m1[(size_t)(b * Nn + j) * Oo + o] + b2;
      lds[m * 128 + o] = x > 0.f ? x : 0.f;
    }
  }
  __syncthreads();

  // MLP layer 1
  f32x8 acc2[8];
#pragma unroll
  for (int nt = 0; nt < 8; ++nt) acc2[nt] = (f32x8)0.0f;
#pragma unroll
  for (int ks = 0; ks < 4; ++ks) {
    h16 a = loadA_f32(&lds[n * 128 + ks * 32 + koff]);
#pragma unroll
    for (int nt = 0; nt < 8; ++nt)
      acc2[nt] = wmma16(a, loadB_h(Wmlp1T + (nt * 16 + n) * Oo + ks * 32 + koff), acc2[nt]);
  }
  __syncthreads();
#pragma unroll
  for (int nt = 0; nt < 8; ++nt) {
    int o = nt * 16 + n;
#pragma unroll
    for (int v = 0; v < 8; ++v) {
      int m = v + hs * 8;
      float x = acc2[nt][v];
      lds[m * 128 + o] = x > 0.f ? x : 0.f;
    }
  }
  __syncthreads();

  // MLP layer 2
  f32x8 acc3[8];
#pragma unroll
  for (int nt = 0; nt < 8; ++nt) acc3[nt] = (f32x8)0.0f;
#pragma unroll
  for (int ks = 0; ks < 4; ++ks) {
    h16 a = loadA_f32(&lds[n * 128 + ks * 32 + koff]);
#pragma unroll
    for (int nt = 0; nt < 8; ++nt)
      acc3[nt] = wmma16(a, loadB_h(Wmlp2T + (nt * 16 + n) * Oo + ks * 32 + koff), acc3[nt]);
  }

  // mask -> NEG, then max over i via native f32 max atomics into msgs_ws[b][j][o]
#pragma unroll
  for (int nt = 0; nt < 8; ++nt) {
    int o = nt * 16 + n;
#pragma unroll
    for (int v = 0; v < 8; ++v) {
      int m = v + hs * 8, j = j0 + m;
      float val = mv[v] ? acc3[nt][v] : NEGV;
      atomicMaxF(&msgs_ws[(size_t)(b * Nn + j) * Oo + o], val);
    }
  }
}

// ---------------- K4: triplet max-plus contraction + @WU3 ----------------
__global__ __launch_bounds__(128) void k_tri(
    const float* t1, const float* t2, const float* t3,
    const float* te1, const float* te2, const float* te3, const float* tg,
    const unsigned char* mask, const float* WU3, float* out_tri) {
  __shared__ float smx[128 * 8];
  int bj = blockIdx.x;  // b*N + j
  int b = bj >> 7, j = bj & 127;
  int k = threadIdx.x;
  bool mjk = (mask[b * Nn + j] | mask[b * Nn + k]) != 0;
  float outer[8], mx[8];
#pragma unroll
  for (int h = 0; h < 8; ++h) {
    outer[h] = t2[(b * Nn + j) * 8 + h] + t3[(b * Nn + k) * 8 + h] +
               te3[((size_t)(b * Nn + j) * Nn + k) * 8 + h] + tg[b * 8 + h];
    mx[h] = -3.0e38f;
  }
  for (int i = 0; i < Nn; ++i) {
    bool valid = mjk || (mask[b * Nn + i] != 0);
    const float* p1 = t1 + (b * Nn + i) * 8;
    const float* q1 = te1 + ((size_t)(b * Nn + i) * Nn + j) * 8;
    const float* q2 = te2 + ((size_t)(b * Nn + i) * Nn + k) * 8;
#pragma unroll
    for (int h = 0; h < 8; ++h) {
      float c = valid ? (p1[h] + q1[h] + q2[h] + outer[h]) : NEGV;
      mx[h] = fmaxf(mx[h], c);
    }
  }
#pragma unroll
  for (int h = 0; h < 8; ++h) smx[k * 8 + h] = mx[h];
  __syncthreads();
  // output phase: thread = output column o = k, coalesced stores
  float wcol[8];
#pragma unroll
  for (int h = 0; h < 8; ++h) wcol[h] = WU3[h * Oo + k];
  float* outp = out_tri + ((size_t)(b * Nn + j)) * Nn * Oo;
  for (int kk = 0; kk < Nn; ++kk) {
    float s = 0.f;
#pragma unroll
    for (int h = 0; h < 8; ++h) s += smx[kk * 8 + h] * wcol[h];
    outp[(size_t)kk * Oo + k] = s;
  }
}

// ---------------- K5: ret = z@WU1 + msgs@WU2 (WMMA), copy msgs out ----------
__global__ __launch_bounds__(32) void k_final(
    const float* z, const float* msgs_ws,
    const _Float16* WU1T, const _Float16* WU2T,
    float* out_ret, float* out_msgs) {
  int tile = blockIdx.x;  // 32 tiles of 16 rows over B*N=512
  int lane = threadIdx.x, n = lane & 15, hs = lane >> 4, koff = hs * 8;
  const float* zrow = z + ((size_t)(tile * 16 + n)) * Cc;
  const float* mrow = msgs_ws + ((size_t)(tile * 16 + n)) * Oo;
  __builtin_prefetch(zrow, 0, 3);
  f32x8 acc[8];
#pragma unroll
  for (int nt = 0; nt < 8; ++nt) acc[nt] = (f32x8)0.0f;
#pragma unroll
  for (int ks = 0; ks < 8; ++ks) {  // z @ WU1 (K=256)
    h16 a = loadA_f32(zrow + ks * 32 + koff);
#pragma unroll
    for (int nt = 0; nt < 8; ++nt)
      acc[nt] = wmma16(a, loadB_h(WU1T + (nt * 16 + n) * Cc + ks * 32 + koff), acc[nt]);
  }
#pragma unroll
  for (int ks = 0; ks < 4; ++ks) {  // msgs @ WU2 (K=128)
    h16 a = loadA_f32(mrow + ks * 32 + koff);
#pragma unroll
    for (int nt = 0; nt < 8; ++nt)
      acc[nt] = wmma16(a, loadB_h(WU2T + (nt * 16 + n) * Oo + ks * 32 + koff), acc[nt]);
  }
#pragma unroll
  for (int nt = 0; nt < 8; ++nt)
#pragma unroll
    for (int v = 0; v < 8; ++v) {
      int r = tile * 16 + v + hs * 8, o = nt * 16 + n;
      out_ret[(size_t)r * Oo + o] = acc[nt][v];
    }
  // copy msgs rows of this tile to output region
  for (int idx = lane; idx < 16 * Oo; idx += 32)
    out_msgs[(size_t)tile * 16 * Oo + idx] = msgs_ws[(size_t)tile * 16 * Oo + idx];
}

// ---------------- workspace layout (bytes) ----------------
constexpr size_t OFF_WmeT   = 0;         // 32768
constexpr size_t OFF_Wmlp1T = 32768;     // 32768
constexpr size_t OFF_Wmlp2T = 65536;     // 32768
constexpr size_t OFF_WU2T   = 98304;     // 32768
constexpr size_t OFF_WteCT  = 131072;    // 8192
constexpr size_t OFF_Wm1T   = 139264;    // 65536
constexpr size_t OFF_Wm2T   = 204800;    // 65536
constexpr size_t OFF_WU1T   = 270336;    // 65536
constexpr size_t OFF_T1     = 335872;    // 16384
constexpr size_t OFF_T2     = 352256;    // 16384
constexpr size_t OFF_T3     = 368640;    // 16384
constexpr size_t OFF_TG     = 385024;    // 256 (128 used)
constexpr size_t OFF_MG     = 385280;    // 2048
constexpr size_t OFF_M1     = 387328;    // 262144
constexpr size_t OFF_M2     = 649472;    // 262144
constexpr size_t OFF_MSGS   = 911616;    // 262144
constexpr size_t OFF_TE1    = 1173760;   // 2097152
constexpr size_t OFF_TE2    = 3270912;   // 2097152
constexpr size_t OFF_TE3    = 5368064;   // 2097152 -> total 7465216

extern "C" void kernel_launch(void* const* d_in, const int* in_sizes, int n_in,
                              void* d_out, int out_size, void* d_ws, size_t ws_size,
                              hipStream_t stream) {
  const float* z    = (const float*)d_in[0];
  const float* e    = (const float*)d_in[1];
  const float* g    = (const float*)d_in[2];
  const unsigned char* mask  = (const unsigned char*)d_in[3];
  const unsigned char* mmask = (const unsigned char*)d_in[4];
  const float* Wt1  = (const float*)d_in[5];
  const float* Wt2  = (const float*)d_in[6];
  const float* Wt3  = (const float*)d_in[7];
  const float* Wte1 = (const float*)d_in[8];
  const float* Wte2 = (const float*)d_in[9];
  const float* Wte3 = (const float*)d_in[10];
  const float* Wtg  = (const float*)d_in[11];
  const float* Wm1  = (const float*)d_in[12];
  const float* Wm2  = (const float*)d_in[13];
  const float* Wme  = (const float*)d_in[14];
  const float* Wmg  = (const float*)d_in[15];
  const float* Wmlp1 = (const float*)d_in[16];
  const float* Wmlp2 = (const float*)d_in[17];
  const float* WU1  = (const float*)d_in[18];
  const float* WU2  = (const float*)d_in[19];
  const float* WU3  = (const float*)d_in[20];

  char* ws = (char*)d_ws;
  _Float16* WmeT   = (_Float16*)(ws + OFF_WmeT);
  _Float16* Wmlp1T = (_Float16*)(ws + OFF_Wmlp1T);
  _Float16* Wmlp2T = (_Float16*)(ws + OFF_Wmlp2T);
  _Float16* WU2T   = (_Float16*)(ws + OFF_WU2T);
  _Float16* WteCT  = (_Float16*)(ws + OFF_WteCT);
  _Float16* Wm1T   = (_Float16*)(ws + OFF_Wm1T);
  _Float16* Wm2T   = (_Float16*)(ws + OFF_Wm2T);
  _Float16* WU1T   = (_Float16*)(ws + OFF_WU1T);
  float* t1   = (float*)(ws + OFF_T1);
  float* t2   = (float*)(ws + OFF_T2);
  float* t3   = (float*)(ws + OFF_T3);
  float* tg   = (float*)(ws + OFF_TG);
  float* mg   = (float*)(ws + OFF_MG);
  float* m1   = (float*)(ws + OFF_M1);
  float* m2   = (float*)(ws + OFF_M2);
  float* msgs = (float*)(ws + OFF_MSGS);
  float* te1  = (float*)(ws + OFF_TE1);
  float* te2  = (float*)(ws + OFF_TE2);
  float* te3  = (float*)(ws + OFF_TE3);

  float* out_ret  = (float*)d_out;                    // (B,N,O)   65536
  float* out_msgs = out_ret + Bb * Nn * Oo;           // (B,N,O)   65536
  float* out_tri  = out_msgs + Bb * Nn * Oo;          // (B,N,N,O) 8388608

  // K0: weights + msgs init (233472 jobs)
  k_prep<<<912, 256, 0, stream>>>(Wme, Wmlp1, Wmlp2, WU2, Wm1, Wm2, WU1,
                                  Wte1, Wte2, Wte3,
                                  WmeT, Wmlp1T, Wmlp2T, WU2T, Wm1T, Wm2T, WU1T,
                                  WteCT, msgs);
  // K1: small scalar dots (12832 jobs)
  k_small<<<51, 256, 0, stream>>>(z, g, mask, Wt1, Wt2, Wt3, Wtg, Wmg,
                                  t1, t2, t3, tg, mg);
  // K2: m1/m2 WMMA GEMMs
  k_m12<<<64, 32, 0, stream>>>(z, mask, Wm1T, Wm2T, m1, m2);
  // K3: fused te + pairwise message MLP + max-reduce (TDM e-staging)
  k_pair<<<Bb * Nn * (Nn / 16), 32, 0, stream>>>(e, mmask, WmeT, Wmlp1T, Wmlp2T,
                                                 WteCT, m1, m2, mg,
                                                 te1, te2, te3, msgs);
  // K4: triplet max-plus + @WU3
  k_tri<<<Bb * Nn, 128, 0, stream>>>(t1, t2, t3, te1, te2, te3, tg, mask, WU3,
                                     out_tri);
  // K5: ret + msgs copy-out
  k_final<<<32, 32, 0, stream>>>(z, msgs, WU1T, WU2T, out_ret, out_msgs);
}
